// UWMRMoE_85882166051157
// MI455X (gfx1250) — matrix operand order repaired and checked by reading, not compiled, not run on hardware
//
#include <hip/hip_runtime.h>

#define DEV_INLINE __device__ __forceinline__

typedef __attribute__((ext_vector_type(16))) __bf16 v16bf;
typedef __attribute__((ext_vector_type(8)))  float  v8f;

// 32-byte fragment viewed as two uint4 (for ds_load_b128) or v16bf (for WMMA)
union Frag { uint4 u[2]; v16bf v; };

constexpr int kN = 8192;    // B*T tokens
constexpr int kD = 1024;    // hidden
constexpr int kE = 8;       // experts
constexpr int kM = 2048;    // ffn dim
constexpr float kLoadCoeff = 0.01f;

// LDS strides (in uints). 20 uints = 80B: 16B-aligned rows, non-pow2 bank spread.
#define LDA_U 20   // A: per-row, 16 uints of data (32 bf16) + 4 pad
#define LDB_U 20   // B: per-column, 16 uints of data (32 K bf16) + 4 pad

// ---------------- helpers ----------------
DEV_INLINE unsigned int pack_bf16x2(float lo, float hi) {
#if __has_builtin(__builtin_amdgcn_cvt_pk_bf16_f32)
  typedef __attribute__((ext_vector_type(2))) __bf16 v2bf;
  union { v2bf v; unsigned int u; } c;
  c.v = __builtin_amdgcn_cvt_pk_bf16_f32(lo, hi);
  return c.u;
#else
  union { __bf16 h[2]; unsigned int u; } c;
  c.h[0] = (__bf16)lo; c.h[1] = (__bf16)hi;
  return c.u;
#endif
}
DEV_INLINE unsigned short f32_bf16(float f) {
  union { __bf16 h; unsigned short s; } c;
  c.h = (__bf16)f;
  return c.s;
}

// CDNA5 async copy: global -> LDS, 16B per lane, tracked on ASYNCcnt.
DEV_INLINE void async_copy_b128(unsigned lds_off, const void* gaddr) {
  asm volatile("global_load_async_to_lds_b128 %0, %1, off"
               :: "v"(lds_off), "v"(gaddr)
               : "memory");
}
DEV_INLINE void wait_async0() {
  asm volatile("s_wait_asynccnt 0x0" ::: "memory");
}

// ---------------- tiny kernels ----------------
__global__ void moe_init_kernel(float* probsum) {
  if (threadIdx.x < kE) probsum[threadIdx.x] = 0.0f;
}

__global__ __launch_bounds__(256)
void moe_router_kernel(const float* __restrict__ x, const float* __restrict__ U,
                       const float* __restrict__ Wr, const float* __restrict__ sb,
                       const float* __restrict__ gb, float* __restrict__ combine,
                       float* __restrict__ probsum) {
  __shared__ float ps[kE];
  int tid = threadIdx.x;
  if (tid < kE) ps[tid] = 0.0f;
  __syncthreads();

  int n = blockIdx.x * blockDim.x + tid;
  float acc[kE];
#pragma unroll
  for (int e = 0; e < kE; ++e) acc[e] = 0.0f;
  const float* xr = x + (size_t)n * kD;
  for (int d = 0; d < kD; ++d) {
    float xv = xr[d];
    const float* wr = Wr + (size_t)d * kE;
#pragma unroll
    for (int e = 0; e < kE; ++e) acc[e] = fmaf(xv, wr[e], acc[e]);
  }
  float u = fminf(fmaxf(U[n], 0.0f), 1.0f);
#pragma unroll
  for (int e = 0; e < kE; ++e) acc[e] += u * sb[e] + (1.0f - u) * gb[e];

  float mx = acc[0];
#pragma unroll
  for (int e = 1; e < kE; ++e) mx = fmaxf(mx, acc[e]);
  float sum = 0.0f, p[kE];
#pragma unroll
  for (int e = 0; e < kE; ++e) { p[e] = __expf(acc[e] - mx); sum += p[e]; }
  float inv = 1.0f / sum;
#pragma unroll
  for (int e = 0; e < kE; ++e) p[e] *= inv;

  // top-2 (ties -> lower index, matching lax.top_k)
  int i1 = 0;
#pragma unroll
  for (int e = 1; e < kE; ++e) if (p[e] > p[i1]) i1 = e;
  int i2 = (i1 == 0) ? 1 : 0;
#pragma unroll
  for (int e = 0; e < kE; ++e) if (e != i1 && p[e] > p[i2]) i2 = e;

  float* cw = combine + (size_t)n * kE;
#pragma unroll
  for (int e = 0; e < kE; ++e)
    cw[e] = (e == i1) ? p[i1] : ((e == i2) ? p[i2] : 0.0f);

#pragma unroll
  for (int e = 0; e < kE; ++e) atomicAdd(&ps[e], p[e]);
  __syncthreads();
  if (tid < kE) atomicAdd(&probsum[tid], ps[tid]);
}

__global__ __launch_bounds__(256)
void moe_cast_x_kernel(const float* __restrict__ x, unsigned short* __restrict__ Xb) {
  size_t gid = (size_t)blockIdx.x * blockDim.x + threadIdx.x;   // one float4 each
  float4 v = ((const float4*)x)[gid];
  uint2 o; o.x = pack_bf16x2(v.x, v.y); o.y = pack_bf16x2(v.z, v.w);
  ((uint2*)Xb)[gid] = o;
}

__global__ void moe_loss_kernel(const float* __restrict__ probsum, float* __restrict__ loss) {
  float s = 0.0f;
#pragma unroll
  for (int e = 0; e < kE; ++e) {
    float m = probsum[e] * (1.0f / (float)kN);
    s += m * m;
  }
  loss[0] = (float)kE * s * kLoadCoeff;
}

// ---------------- WMMA GEMM kernels ----------------
// Block tile: 128 rows x 64 cols, 256 threads = 8 waves, wave w -> rows [16w,16w+16),
// each wave does 4 WMMA n-tiles of 16 cols. K staged 32 (bf16) per step through LDS.
//
// LDS A layout (fragment-friendly): per row, the four K-groups of 8 bf16 are stored
// in order [K0-7, K8-15, K16-23, K24-31], so lane-half h reads its 16 A elements
// (K = 16h + {0..15}) as one contiguous 32B chunk -> 2x ds_load_b128.
// A tile is staged with global_load_async_to_lds_b128 (ASYNCcnt), K-group order
// kept as in global memory (no permutation).
// LDS B layout: column-major (K contiguous per column), so a lane's 16 B elements
// (K = 16h + {0..15}) for its column are contiguous -> 2x ds_load_b128.

DEV_INLINE void load_frag(Frag& f, const unsigned int* base) {
  f.u[0] = *(const uint4*)(base);
  f.u[1] = *(const uint4*)(base + 4);
}

// H = silu(Xb@Wg) * (Xb@Wu) * combine  -> bf16
__global__ __launch_bounds__(256)
void moe_gateup_wmma(const unsigned short* __restrict__ Xb,   // [N,D] bf16
                     const float* __restrict__ Wg,            // [D,M] fp32
                     const float* __restrict__ Wu,            // [D,M] fp32
                     const float* __restrict__ cscale,        // combine+e (stride E) or null
                     unsigned short* __restrict__ Hb)         // [N,M] bf16
{
  __shared__ unsigned int AsU[128 * LDA_U];   // 10.0 KB
  __shared__ unsigned int BgU[64 * LDB_U];    //  5.0 KB (column-major)
  __shared__ unsigned int BuU[64 * LDB_U];    //  5.0 KB

  const int tid  = threadIdx.x;
  const int wave = tid >> 5;
  const int lane = tid & 31;
  const int laneRow = lane & 15;
  const int laneHi  = lane >> 4;
  const int row0 = blockIdx.y * 128;
  const int col0 = blockIdx.x * 64;

  v8f accG[4], accU[4];
  v8f z = {0.f,0.f,0.f,0.f,0.f,0.f,0.f,0.f};
#pragma unroll
  for (int j = 0; j < 4; ++j) { accG[j] = z; accU[j] = z; }

  // B staging coords: each thread handles 2 consecutive K rows x 4 columns
  const int r2 = (tid >> 4) * 2;        // K row pair base: 0,2,..,30
  const int cq = (tid & 15) * 4;        // column quad base: 0,4,..,60
  const int ku = r2 >> 1;               // uint index along K within a column

  // Per-thread async A-copy coordinates (2 jobs of one uint4 = one K-group each)
  unsigned aLds[2];
  const unsigned short* aGlb[2];
#pragma unroll
  for (int it = 0; it < 2; ++it) {
    int u4 = tid + it * 256;                  // 0..511
    int r  = u4 >> 2;                         // row 0..127
    int g  = u4 & 3;                          // K-group 0..3
    int gp = g;                               // identity (contiguous K)
    aLds[it] = (unsigned)(unsigned long long)&AsU[r * LDA_U + gp * 4];
    aGlb[it] = Xb + (size_t)(row0 + r) * kD + g * 8;   // + k0 added in loop
  }

  for (int k0 = 0; k0 < kD; k0 += 32) {
    // ---- stage A asynchronously: 128x32 bf16 via async b128 copies ----
#pragma unroll
    for (int it = 0; it < 2; ++it)
      async_copy_b128(aLds[it], (const void*)(aGlb[it] + k0));
    // ---- stage Bg/Bu: 32x64 fp32 -> bf16, transposed to column-major ----
    {
      const float* wg0 = Wg + (size_t)(k0 + r2) * kM + col0 + cq;
      const float* wu0 = Wu + (size_t)(k0 + r2) * kM + col0 + cq;
      float4 g0 = *(const float4*)(wg0);
      float4 g1 = *(const float4*)(wg0 + kM);
      float4 u0 = *(const float4*)(wu0);
      float4 u1 = *(const float4*)(wu0 + kM);
      BgU[(cq + 0) * LDB_U + ku] = pack_bf16x2(g0.x, g1.x);
      BgU[(cq + 1) * LDB_U + ku] = pack_bf16x2(g0.y, g1.y);
      BgU[(cq + 2) * LDB_U + ku] = pack_bf16x2(g0.z, g1.z);
      BgU[(cq + 3) * LDB_U + ku] = pack_bf16x2(g0.w, g1.w);
      BuU[(cq + 0) * LDB_U + ku] = pack_bf16x2(u0.x, u1.x);
      BuU[(cq + 1) * LDB_U + ku] = pack_bf16x2(u0.y, u1.y);
      BuU[(cq + 2) * LDB_U + ku] = pack_bf16x2(u0.z, u1.z);
      BuU[(cq + 3) * LDB_U + ku] = pack_bf16x2(u0.w, u1.w);
      if (k0 + 32 < kD) {                      // prefetch next K-step weight rows
        __builtin_prefetch(wg0 + (size_t)32 * kM, 0, 0);
        __builtin_prefetch(wu0 + (size_t)32 * kM, 0, 0);
      }
    }
    wait_async0();                             // A-tile async copies landed in LDS
    __syncthreads();

    // ---- compute ----
    Frag a;
    const int arow = wave * 16 + laneRow;
    load_frag(a, &AsU[arow * LDA_U + laneHi * 8]);       // 2x ds_load_b128
#pragma unroll
    for (int j = 0; j < 4; ++j) {
      const int bcol = j * 16 + laneRow;
      Frag bg, bu;
      load_frag(bg, &BgU[bcol * LDB_U + laneHi * 8]);
      load_frag(bu, &BuU[bcol * LDB_U + laneHi * 8]);
      accG[j] = __builtin_amdgcn_wmma_f32_16x16x32_bf16(false, a.v, false, bg.v,
                                                        (short)0, accG[j], false, false);
      accU[j] = __builtin_amdgcn_wmma_f32_16x16x32_bf16(false, a.v, false, bu.v,
                                                        (short)0, accU[j], false, false);
    }
    __syncthreads();
  }

  // ---- epilogue: silu(g)*u*combine -> bf16 ----
  float cs[8];
#pragma unroll
  for (int r = 0; r < 8; ++r) {
    int rowG = row0 + wave * 16 + r + laneHi * 8;
    cs[r] = cscale ? cscale[(size_t)rowG * kE] : 1.0f;
  }
#pragma unroll
  for (int j = 0; j < 4; ++j) {
    int colG = col0 + j * 16 + laneRow;
#pragma unroll
    for (int r = 0; r < 8; ++r) {
      int rowG = row0 + wave * 16 + r + laneHi * 8;
      float g = accG[j][r];
      float u = accU[j][r];
      float h = (g / (1.0f + __expf(-g))) * u * cs[r];
      Hb[(size_t)rowG * kM + colG] = f32_bf16(h);
    }
  }
}

// out (+)= Hb @ Wd
__global__ __launch_bounds__(256)
void moe_down_wmma(const unsigned short* __restrict__ Hb,    // [N,M] bf16
                   const float* __restrict__ Wd,             // [M,D] fp32
                   float* __restrict__ out,                  // [N,D] fp32
                   int accum)
{
  __shared__ unsigned int AsU[128 * LDA_U];
  __shared__ unsigned int BdU[64 * LDB_U];

  const int tid  = threadIdx.x;
  const int wave = tid >> 5;
  const int lane = tid & 31;
  const int laneRow = lane & 15;
  const int laneHi  = lane >> 4;
  const int row0 = blockIdx.y * 128;
  const int col0 = blockIdx.x * 64;

  v8f acc[4];
  v8f z = {0.f,0.f,0.f,0.f,0.f,0.f,0.f,0.f};
#pragma unroll
  for (int j = 0; j < 4; ++j) acc[j] = z;

  const int r2 = (tid >> 4) * 2;
  const int cq = (tid & 15) * 4;
  const int ku = r2 >> 1;

  unsigned aLds[2];
  const unsigned short* aGlb[2];
#pragma unroll
  for (int it = 0; it < 2; ++it) {
    int u4 = tid + it * 256;
    int r  = u4 >> 2;
    int g  = u4 & 3;
    int gp = g;
    aLds[it] = (unsigned)(unsigned long long)&AsU[r * LDA_U + gp * 4];
    aGlb[it] = Hb + (size_t)(row0 + r) * kM + g * 8;
  }

  for (int k0 = 0; k0 < kM; k0 += 32) {
#pragma unroll
    for (int it = 0; it < 2; ++it)
      async_copy_b128(aLds[it], (const void*)(aGlb[it] + k0));
    {
      const float* wd0 = Wd + (size_t)(k0 + r2) * kD + col0 + cq;
      float4 d0 = *(const float4*)(wd0);
      float4 d1 = *(const float4*)(wd0 + kD);
      BdU[(cq + 0) * LDB_U + ku] = pack_bf16x2(d0.x, d1.x);
      BdU[(cq + 1) * LDB_U + ku] = pack_bf16x2(d0.y, d1.y);
      BdU[(cq + 2) * LDB_U + ku] = pack_bf16x2(d0.z, d1.z);
      BdU[(cq + 3) * LDB_U + ku] = pack_bf16x2(d0.w, d1.w);
      if (k0 + 32 < kM) __builtin_prefetch(wd0 + (size_t)32 * kD, 0, 0);
    }
    wait_async0();
    __syncthreads();

    Frag a;
    const int arow = wave * 16 + laneRow;
    load_frag(a, &AsU[arow * LDA_U + laneHi * 8]);
#pragma unroll
    for (int j = 0; j < 4; ++j) {
      const int bcol = j * 16 + laneRow;
      Frag b;
      load_frag(b, &BdU[bcol * LDB_U + laneHi * 8]);
      acc[j] = __builtin_amdgcn_wmma_f32_16x16x32_bf16(false, a.v, false, b.v,
                                                       (short)0, acc[j], false, false);
    }
    __syncthreads();
  }

#pragma unroll
  for (int j = 0; j < 4; ++j) {
    int colG = col0 + j * 16 + laneRow;
#pragma unroll
    for (int r = 0; r < 8; ++r) {
      int rowG = row0 + wave * 16 + r + laneHi * 8;
      size_t idx = (size_t)rowG * kD + colG;
      float v = acc[j][r];
      if (accum) out[idx] += v;
      else       out[idx] = v;
    }
  }
}

// ---------------- launcher ----------------
extern "C" void kernel_launch(void* const* d_in, const int* in_sizes, int n_in,
                              void* d_out, int out_size, void* d_ws, size_t ws_size,
                              hipStream_t stream) {
  (void)in_sizes; (void)n_in; (void)out_size; (void)ws_size;
  const float* x  = (const float*)d_in[0];
  const float* U  = (const float*)d_in[1];
  const float* Wr = (const float*)d_in[2];
  const float* sb = (const float*)d_in[3];
  const float* gb = (const float*)d_in[4];
  const float* Wg = (const float*)d_in[5];
  const float* Wu = (const float*)d_in[6];
  const float* Wd = (const float*)d_in[7];
  const float* Sg = (const float*)d_in[8];
  const float* Su = (const float*)d_in[9];
  const float* Sd = (const float*)d_in[10];
  float* out = (float*)d_out;

  // workspace layout (~50.6 MB)
  char* ws = (char*)d_ws;
  unsigned short* Xb = (unsigned short*)ws;                              // N*D bf16
  unsigned short* Hb = (unsigned short*)(ws + (size_t)kN * kD * 2);      // N*M bf16
  float* combine = (float*)(ws + (size_t)kN * kD * 2 + (size_t)kN * kM * 2); // N*E f32
  float* probsum = combine + (size_t)kN * kE;                            // E f32

  hipLaunchKernelGGL(moe_init_kernel, dim3(1), dim3(32), 0, stream, probsum);
  hipLaunchKernelGGL(moe_router_kernel, dim3(kN / 256), dim3(256), 0, stream,
                     x, U, Wr, sb, gb, combine, probsum);
  hipLaunchKernelGGL(moe_cast_x_kernel, dim3((kN * kD / 4) / 256), dim3(256), 0, stream,
                     x, Xb);

  dim3 blk(256);
  dim3 guGrid(kM / 64, kN / 128);
  dim3 dnGrid(kD / 64, kN / 128);

  // shared expert first: writes out (accum=0)
  hipLaunchKernelGGL(moe_gateup_wmma, guGrid, blk, 0, stream, Xb, Sg, Su,
                     (const float*)nullptr, Hb);
  hipLaunchKernelGGL(moe_down_wmma, dnGrid, blk, 0, stream, Hb, Sd, out, 0);

  // routed experts accumulate
  for (int e = 0; e < kE; ++e) {
    hipLaunchKernelGGL(moe_gateup_wmma, guGrid, blk, 0, stream,
                       Xb, Wg + (size_t)e * kD * kM, Wu + (size_t)e * kD * kM,
                       combine + e, Hb);
    hipLaunchKernelGGL(moe_down_wmma, dnGrid, blk, 0, stream,
                       Hb, Wd + (size_t)e * kM * kD, out, 1);
  }

  hipLaunchKernelGGL(moe_loss_kernel, dim3(1), dim3(1), 0, stream,
                     probsum, out + (size_t)kN * kD);
}
